// Model_60378650247539
// MI455X (gfx1250) — matrix operand order repaired and checked by reading, not compiled, not run on hardware
//
#include <hip/hip_runtime.h>
#include <stdint.h>

// Problem constants
#define B_ 128
#define H_ 512
#define T_ 512
#define G_ 2048
#define NWG 32   // persistent workgroups; 32 WGs x 8 waves = 256 waves = 8 Mtiles x 32 Jtiles

typedef __attribute__((ext_vector_type(16))) __bf16   v16bf;
typedef __attribute__((ext_vector_type(8)))  float    v8f;
typedef __attribute__((ext_vector_type(4)))  unsigned v4u;
typedef __attribute__((ext_vector_type(4)))  int      v4i;

union FragAB { v16bf v; v4u q[2]; };

#if __has_builtin(__builtin_amdgcn_global_load_async_to_lds_b128)
#define HAVE_ASYNC_LDS 1
#else
#define HAVE_ASYNC_LDS 0
#endif

__device__ __forceinline__ void wait_asynccnt0() {
#if __has_builtin(__builtin_amdgcn_s_wait_asynccnt)
  __builtin_amdgcn_s_wait_asynccnt(0);
#else
  asm volatile("s_wait_asynccnt 0x0" ::: "memory");
#endif
}

// Stage 16 bytes per lane from global into LDS (async on CDNA5).
__device__ __forceinline__ void stage16(const void* g, void* l) {
#if HAVE_ASYNC_LDS
  v4i* gt = (v4i*)g;   // C-style cast drops const; builtin param is non-const
  v4i* lt = (v4i*)l;
  __builtin_amdgcn_global_load_async_to_lds_b128(
      (__attribute__((address_space(1))) v4i*)gt,
      (__attribute__((address_space(3))) v4i*)lt, 0, 0);
#else
  *(v4u*)l = *(const v4u*)g;
#endif
}

__device__ __forceinline__ unsigned short f2bf(float f) {
  unsigned u = __float_as_uint(f);
  u += 0x7FFFu + ((u >> 16) & 1u);       // round-to-nearest-even
  return (unsigned short)(u >> 16);
}
__device__ __forceinline__ float sigf(float x) { return 1.f / (1.f + __expf(-x)); }
__device__ __forceinline__ float tanhf_(float x) {
  x = fminf(15.f, fmaxf(-15.f, x));
  float e = __expf(2.f * x);
  return (e - 1.f) / (e + 1.f);
}
__device__ __forceinline__ unsigned hash32(unsigned x) {
  x ^= x >> 16; x *= 0x7feb352du; x ^= x >> 15; x *= 0x846ca68bu; x ^= x >> 16;
  return x;
}

// Monotonic-counter grid barrier (all NWG workgroups are co-resident).
__device__ __forceinline__ void grid_barrier(unsigned* cnt) {
  __threadfence();
  __syncthreads();
  if (threadIdx.x == 0) {
    unsigned old = __hip_atomic_fetch_add(cnt, 1u, __ATOMIC_ACQ_REL, __HIP_MEMORY_SCOPE_AGENT);
    unsigned target = (old / NWG + 1u) * NWG;
    while (__hip_atomic_load(cnt, __ATOMIC_ACQUIRE, __HIP_MEMORY_SCOPE_AGENT) < target)
      __builtin_amdgcn_s_sleep(2);
  }
  __syncthreads();
}

// ---------------------------------------------------------------------------
// Prep kernel 1: zero barrier counter, h ping-pong state, and output.
__global__ void init_ws(unsigned* counter, unsigned* h32, float* out) {
  int i = blockIdx.x * blockDim.x + threadIdx.x;
  if (i < 256) counter[i] = 0u;
  if (i < (2 * 4 * B_ * H_ / 2)) h32[i] = 0u;  // 2 parities * 4 layers * 128*512 bf16
  if (i < B_ * T_) out[i] = 0.f;
}

// Prep kernel 2: fp32 -> bf16 + shuffle into per-lane WMMA B-fragment layout.
// dst layout: [kt][ntile][lane 0..31][16 bf16], tile = 32x16 (KxN).
// Lane L holds column n = ntile*16 + (L&15), k = kt*32 + 16*(L>>4) + p  (p=0..15).
// W[k][n] = (k < ksplit) ? w_ih[n][k] : w_hh[n][k-ksplit]
__global__ void shuffle_w(const float* w_ih, const float* w_hh,
                          unsigned short* dst, int nkt, int ksplit) {
  long e = (long)blockIdx.x * blockDim.x + threadIdx.x;
  long total = (long)nkt * 128 * 512;
  if (e >= total) return;
  int  within = (int)(e & 511);
  long tile   = e >> 9;
  int  lane = within >> 4, p = within & 15;
  int  kt = (int)(tile / 128), nt = (int)(tile % 128);
  int  k = kt * 32 + ((lane >> 4) * 16) + p;
  int  n = nt * 16 + (lane & 15);
  float val = (k < ksplit) ? w_ih[(size_t)n * 512 + k]
                           : w_hh[(size_t)n * 512 + (k - ksplit)];
  dst[e] = f2bf(val);
}

// ---------------------------------------------------------------------------
struct MainParams {
  const float* x;        // [B,T]
  const float* w_ih1;    // [2048] (layer-0 input weight, scalar input)
  const float* bih[4];
  const float* bhh[4];
  const float* w_lin;    // [512]
  const float* b_lin;    // [1]
  const unsigned short* W[4];  // shuffled bf16 weights
  unsigned short* hbuf;  // [2 parity][4 layer][128][512] bf16 (post-dropout h)
  unsigned* counter;
  float* out;            // [B,T]
};

__global__ __launch_bounds__(256) void lstm_persistent(MainParams P) {
  // Double-buffered B-tile staging: 2 x (4 gates x 32x16 tile) = 8 KB
  __shared__ __align__(16) unsigned short ldsW[2][4 * 512];

  const int tid  = threadIdx.x;
  const int lane = tid & 31;
  const int wv   = tid >> 5;
  const int gw   = blockIdx.x * 8 + wv;   // all waves of a WG share jt
  const int mt   = gw & 7;                // batch tile 0..7
  const int jt   = gw >> 3;               // hidden-unit tile 0..31
  const int m0   = mt * 16;
  const int half = lane >> 4;
  const int nlo  = lane & 15;
  const int j    = jt * 16 + nlo;         // this lane's hidden unit column

  // Per-lane constants (biases, layer-0 input weights, output weights)
  float bias[4][4], wi1[4];
  #pragma unroll
  for (int l = 0; l < 4; ++l)
    #pragma unroll
    for (int g = 0; g < 4; ++g) {
      int n = g * H_ + j;
      bias[l][g] = P.bih[l][n] + P.bhh[l][n];
    }
  #pragma unroll
  for (int g = 0; g < 4; ++g) wi1[g] = P.w_ih1[g * H_ + j];
  const float wlin_j = P.w_lin[j];
  const float blin   = P.b_lin[0];

  // Cell state for all 4 layers lives in registers for the whole sequence.
  float creg[4][8];
  #pragma unroll
  for (int l = 0; l < 4; ++l)
    #pragma unroll
    for (int r = 0; r < 8; ++r) creg[l][r] = 0.f;

  const v8f vzero = {0.f, 0.f, 0.f, 0.f, 0.f, 0.f, 0.f, 0.f};
  const int stg_g = tid >> 6;   // gate this thread stages
  const int stg_q = tid & 63;   // 16B chunk within the 1KB gate tile

  for (int t = 0; t < T_; ++t) {
    const int cur = t & 1, prv = cur ^ 1;
    for (int l = 0; l < 4; ++l) {
      const unsigned short* Wl  = P.W[l];
      unsigned short*       hc  = P.hbuf + (size_t)(cur * 4 + l) * (B_ * H_);
      const unsigned short* hp  = P.hbuf + (size_t)(prv * 4 + l) * (B_ * H_);
      const unsigned short* hlo = (l > 0) ? P.hbuf + (size_t)(cur * 4 + (l - 1)) * (B_ * H_) : hp;
      const int nkt = (l == 0) ? 16 : 32;   // K = 512 (recurrent only) or 1024 (concat)

      v8f acc[4];
      #pragma unroll
      for (int g = 0; g < 4; ++g) acc[g] = vzero;

      // Stage k-tile 0 (4 gate tiles, 4KB; 16B per lane, async to LDS)
      stage16(Wl + (size_t)(stg_g * 32 + jt) * 512 + (size_t)stg_q * 8,
              &ldsW[0][(size_t)tid * 8]);
      for (int kt = 0; kt < nkt; ++kt) {
        wait_asynccnt0();    // this wave's staging for buf[kt&1] has landed
        __syncthreads();     // ...and every other wave's too (and reads of other buf done)
        if (kt + 1 < nkt) {  // async-stage next k-tile into the other buffer
          stage16(Wl + (size_t)((kt + 1) * 128 + (stg_g * 32 + jt)) * 512 + (size_t)stg_q * 8,
                  &ldsW[(kt + 1) & 1][(size_t)tid * 8]);
        }
        // A fragment: 16x32 bf16, contiguous k = 16*half + s per lane (same as B) -> two b128 loads
        const unsigned short* Ab;
        int lk;
        if (l == 0)       { Ab = hp;  lk = kt * 32; }
        else if (kt < 16) { Ab = hlo; lk = kt * 32; }           // lower-layer input
        else              { Ab = hp;  lk = (kt - 16) * 32; }    // own recurrent h
        const int m = m0 + nlo;
        const v4u* pa = (const v4u*)(Ab + (size_t)m * H_ + lk + 16 * half);
        FragAB a;
        a.q[0] = pa[0];
        a.q[1] = pa[1];
        __builtin_prefetch((const void*)(pa + 4), 0, 0);  // next k-tile of A

        const v4u* lb = (const v4u*)ldsW[kt & 1];
        #pragma unroll
        for (int g = 0; g < 4; ++g) {
          FragAB b;
          b.q[0] = lb[g * 64 + lane * 2];
          b.q[1] = lb[g * 64 + lane * 2 + 1];
          acc[g] = __builtin_amdgcn_wmma_f32_16x16x32_bf16(
              false, a.v, false, b.v, (short)0, acc[g], false, false);
        }
      }

      // Activations, cell update, dropout, h store (+ projection on top layer)
      #pragma unroll
      for (int r = 0; r < 8; ++r) {
        const int b = m0 + r + 8 * half;   // C-fragment row mapping
        float gi = acc[0][r] + bias[l][0];
        float gf = acc[1][r] + bias[l][1];
        float gg = acc[2][r] + bias[l][2];
        float go = acc[3][r] + bias[l][3];
        if (l == 0) {
          float xv = P.x[(size_t)b * T_ + t];
          gi += xv * wi1[0]; gf += xv * wi1[1]; gg += xv * wi1[2]; go += xv * wi1[3];
        }
        float i_ = sigf(gi), f_ = sigf(gf), g_ = tanhf_(gg), o_ = sigf(go);
        float cn = f_ * creg[l][r] + i_ * g_;
        creg[l][r] = cn;
        float hv = o_ * tanhf_(cn);
        // Bernoulli(0.5) dropout * 2 (deterministic hash in place of threefry)
        unsigned key = (((unsigned)l * T_ + (unsigned)t) * B_ + (unsigned)b) * H_ + (unsigned)j;
        hv *= (hash32(key ^ 0x9E3779B9u) & 1u) ? 2.0f : 0.0f;
        hc[(size_t)b * H_ + j] = f2bf(hv);
        if (l == 3) {  // out[b,t] = sum_j h4drop * w_lin + b_lin
          float red = hv * wlin_j;
          red += __shfl_xor(red, 1);
          red += __shfl_xor(red, 2);
          red += __shfl_xor(red, 4);
          red += __shfl_xor(red, 8);
          if (nlo == 0) {
            float tot = red + (jt == 0 ? blin : 0.f);
            atomicAdd(&P.out[(size_t)b * T_ + t], tot);
          }
        }
      }
      grid_barrier(P.counter);  // publish h[l] before anyone consumes it
    }
  }
}

// ---------------------------------------------------------------------------
extern "C" void kernel_launch(void* const* d_in, const int* in_sizes, int n_in,
                              void* d_out, int out_size, void* d_ws, size_t ws_size,
                              hipStream_t stream) {
  (void)in_sizes; (void)n_in; (void)out_size; (void)ws_size;
  const float* x = (const float*)d_in[0];
  const float *w_ih[4], *w_hh[4], *b_ih[4], *b_hh[4];
  for (int l = 0; l < 4; ++l) {
    w_ih[l] = (const float*)d_in[1 + 4 * l];
    w_hh[l] = (const float*)d_in[2 + 4 * l];
    b_ih[l] = (const float*)d_in[3 + 4 * l];
    b_hh[l] = (const float*)d_in[4 + 4 * l];
  }
  const float* w_lin = (const float*)d_in[17];
  const float* b_lin = (const float*)d_in[18];

  // Workspace layout: [counter 1KB][h ping-pong 1MB][bf16 weights 14MB]
  uint8_t* ws = (uint8_t*)d_ws;
  unsigned* counter = (unsigned*)ws;
  unsigned short* hbuf = (unsigned short*)(ws + 1024);
  unsigned short* W[4];
  W[0] = (unsigned short*)(ws + 1024 + (size_t)2 * 4 * B_ * H_ * 2);
  W[1] = W[0] + (size_t)16 * 128 * 512;
  W[2] = W[1] + (size_t)32 * 128 * 512;
  W[3] = W[2] + (size_t)32 * 128 * 512;

  init_ws<<<1024, 256, 0, stream>>>(counter, (unsigned*)hbuf, (float*)d_out);
  shuffle_w<<<(16 * 128 * 512) / 256, 256, 0, stream>>>(w_hh[0], w_hh[0], W[0], 16, 0);
  for (int l = 1; l < 4; ++l)
    shuffle_w<<<(32 * 128 * 512) / 256, 256, 0, stream>>>(w_ih[l], w_hh[l], W[l], 32, 512);

  MainParams P;
  P.x = x; P.w_ih1 = w_ih[0];
  for (int l = 0; l < 4; ++l) { P.bih[l] = b_ih[l]; P.bhh[l] = b_hh[l]; P.W[l] = W[l]; }
  P.w_lin = w_lin; P.b_lin = b_lin;
  P.hbuf = hbuf; P.counter = counter; P.out = (float*)d_out;
  lstm_persistent<<<NWG, 256, 0, stream>>>(P);
}
